// GraphSpatialIntegration_39522289057932
// MI455X (gfx1250) — matrix operand order repaired and checked by reading, not compiled, not run on hardware
//
#include <hip/hip_runtime.h>
#include <hip/hip_bf16.h>

typedef __attribute__((ext_vector_type(16))) _Float16 v16h;
typedef __attribute__((ext_vector_type(8)))  _Float16 v8h;
typedef __attribute__((ext_vector_type(8)))  float    v8f;

#define WAVES_PER_BLOCK 8
#define BLOCK_THREADS   (WAVES_PER_BLOCK * 32)

// ---------------------------------------------------------------------------
// WMMA helpers (CDNA5 gfx1250, wave32)
// ---------------------------------------------------------------------------
__device__ __forceinline__ v8f wmma_f16(v16h a, v16h b, v8f c) {
  // (neg_a, A, neg_b, B, c_mod, C, reuse_a, reuse_b)
  return __builtin_amdgcn_wmma_f32_16x16x32_f16(false, a, false, b, (short)0, c,
                                                false, false);
}

// A-matrix fragment, 16x32 fp16, per ISA layout:
//   lanes 0-15 : row M = lane,    K = {0..7, 16..23}   (+ kb*32)
//   lanes 16-31: row M = lane-16, K = {8..15, 24..31}  (+ kb*32)
// `base` is row-major [16][rowLen] fp16 in LDS.
__device__ __forceinline__ v16h load_frag_a(const _Float16* base, int rowLen,
                                            int lane, int kb) {
  int m    = lane & 15;
  int half = lane >> 4;  // 0 or 1
  const _Float16* p = base + m * rowLen + kb * 32 + half * 8;
  v8h lo = *(const v8h*)(p);       // K = kb*32 + half*8 + 0..7
  v8h hi = *(const v8h*)(p + 16);  // K = kb*32 + half*8 + 16..23
  return __builtin_shufflevector(lo, hi, 0, 1, 2, 3, 4, 5, 6, 7, 8, 9, 10, 11,
                                 12, 13, 14, 15);
}

// B-matrix fragment, 32x16 fp16, per ISA layout:
//   lanes 0-15 : col N = lane,    K = {0..7, 16..23}  (+ kb*32)
//   lanes 16-31: col N = lane-16, K = {8..15, 24..31} (+ kb*32)
// `baseT` is the weight stored TRANSPOSED in LDS: [N][K] fp16, row length = K.
__device__ __forceinline__ v16h load_frag_b(const _Float16* baseT, int rowLen,
                                            int lane, int kb, int nb) {
  int n      = nb * 16 + (lane & 15);
  int kstart = kb * 32 + (lane >> 4) * 8;
  const _Float16* p = baseT + n * rowLen + kstart;
  v8h lo = *(const v8h*)(p);
  v8h hi = *(const v8h*)(p + 16);
  return __builtin_shufflevector(lo, hi, 0, 1, 2, 3, 4, 5, 6, 7, 8, 9, 10, 11,
                                 12, 13, 14, 15);
}

// Convert two float4 (8 consecutive fp32) into a packed v8h.
__device__ __forceinline__ v8h pack8h(float4 a, float4 b) {
  v8h h;
  h[0] = (_Float16)a.x; h[1] = (_Float16)a.y;
  h[2] = (_Float16)a.z; h[3] = (_Float16)a.w;
  h[4] = (_Float16)b.x; h[5] = (_Float16)b.y;
  h[6] = (_Float16)b.z; h[7] = (_Float16)b.w;
  return h;
}

// Cooperative block staging: weight W [K][N=64] (row-major fp32) -> LDS
// transposed [N][K] fp16, using 16B vector LDS stores (8 K-contiguous halves).
__device__ __forceinline__ void stage_weightT(_Float16* dstT, const float* W,
                                              int K, int tid) {
  int chunks = (64 * K) / 8;
  for (int c = tid; c < chunks; c += BLOCK_THREADS) {
    int i  = c * 8;
    int n  = i / K;        // output row (N)
    int k0 = i - n * K;    // starting K
    v8h h;
#pragma unroll
    for (int j = 0; j < 8; ++j) h[j] = (_Float16)W[(k0 + j) * 64 + n];
    *(v8h*)&dstT[i] = h;
  }
}

// ---------------------------------------------------------------------------
// Kernel 0: zero the aggregation workspace
// ---------------------------------------------------------------------------
__global__ void zero_f32_kernel(float* __restrict__ p, long long n) {
  long long i = (long long)blockIdx.x * blockDim.x + threadIdx.x;
  if (i < n) p[i] = 0.0f;
}

// ---------------------------------------------------------------------------
// Kernel 1: per-edge message MLP + scatter-add aggregation
//   m = ReLU(cat[x[dst], x[src]] @ W1 + b1) @ W2 + b2 ; aggr[dst] += m
// One wave handles tiles of 16 edges (grid-stride, weights stay LDS-resident).
// ---------------------------------------------------------------------------
__global__ __launch_bounds__(BLOCK_THREADS) void edge_msg_kernel(
    const float* __restrict__ x, const int* __restrict__ eidx,
    const float* __restrict__ W1, const float* __restrict__ b1,
    const float* __restrict__ W2, const float* __restrict__ b2,
    float* __restrict__ aggr, int nEdges) {
  __shared__ __align__(32) _Float16 sW1T[64 * 128];  // W1 transposed: [N][K=128]
  __shared__ __align__(32) _Float16 sW2T[64 * 64];   // W2 transposed: [N][K=64]
  __shared__ float sB1[64];
  __shared__ float sB2[64];
  __shared__ __align__(32) _Float16 sMin[WAVES_PER_BLOCK][16 * 128];  // A tile L1
  __shared__ __align__(32) _Float16 sH[WAVES_PER_BLOCK][16 * 64];     // A tile L2
  __shared__ int sDst[WAVES_PER_BLOCK][16];

  const int tid = threadIdx.x;
  stage_weightT(sW1T, W1, 128, tid);
  stage_weightT(sW2T, W2, 64, tid);
  if (tid < 64) {
    sB1[tid] = b1[tid];
    sB2[tid] = b2[tid];
  }
  __syncthreads();

  const int lane = tid & 31;
  const int wid  = tid >> 5;
  const int* __restrict__ srcIdx = eidx;           // edge_index row 0
  const int* __restrict__ dstIdx = eidx + nEdges;  // edge_index row 1

  const int nTiles = (nEdges + 15) >> 4;

  for (int tile = blockIdx.x * WAVES_PER_BLOCK + wid; tile < nTiles;
       tile += gridDim.x * WAVES_PER_BLOCK) {
    const int ebase = tile << 4;

    // ---- stage concat[x[dst], x[src]] tile (16 x 128) fp16, 16B LDS stores -
    {
      int el   = lane >> 1;  // local edge 0..15
      int part = lane & 1;   // 0 -> x_i = x[dst] (cols 0..63), 1 -> x_j = x[src]
      int edge = ebase + el;
      if (edge >= nEdges) edge = nEdges - 1;
      int node = part ? srcIdx[edge] : dstIdx[edge];
      const float4* xr = (const float4*)(x + (long long)node * 64);
      _Float16* drow = &sMin[wid][el * 128 + part * 64];
#pragma unroll
      for (int j = 0; j < 8; ++j) {
        *(v8h*)(drow + j * 8) = pack8h(xr[2 * j], xr[2 * j + 1]);
      }
      if (lane < 16) {
        int e2 = ebase + lane;
        sDst[wid][lane] = dstIdx[e2 < nEdges ? e2 : nEdges - 1];
      }
    }
    asm volatile("s_wait_dscnt 0" ::: "memory");  // within-wave LDS RAW order

    // ---- layer 1: h = ReLU([16x128] @ W1 + b1) ----------------------------
#pragma unroll
    for (int nb = 0; nb < 4; ++nb) {
      float bias = sB1[nb * 16 + (lane & 15)];
      v8f acc = {bias, bias, bias, bias, bias, bias, bias, bias};
#pragma unroll
      for (int kb = 0; kb < 4; ++kb) {
        v16h a = load_frag_a(&sMin[wid][0], 128, lane, kb);
        v16h b = load_frag_b(sW1T, 128, lane, kb, nb);
        acc = wmma_f16(a, b, acc);
      }
      int col = lane & 15, rbase = (lane >> 4) * 8;
#pragma unroll
      for (int r = 0; r < 8; ++r) {
        float v = acc[r] > 0.0f ? acc[r] : 0.0f;
        sH[wid][(rbase + r) * 64 + nb * 16 + col] = (_Float16)v;
      }
    }
    asm volatile("s_wait_dscnt 0" ::: "memory");

    // ---- layer 2: m = h @ W2 + b2 ; scatter-add to aggr[dst] --------------
#pragma unroll
    for (int nb = 0; nb < 4; ++nb) {
      float bias = sB2[nb * 16 + (lane & 15)];
      v8f acc = {bias, bias, bias, bias, bias, bias, bias, bias};
#pragma unroll
      for (int kb = 0; kb < 2; ++kb) {
        v16h a = load_frag_a(&sH[wid][0], 64, lane, kb);
        v16h b = load_frag_b(sW2T, 64, lane, kb, nb);
        acc = wmma_f16(a, b, acc);
      }
      int col = nb * 16 + (lane & 15), rbase = (lane >> 4) * 8;
#pragma unroll
      for (int r = 0; r < 8; ++r) {
        int row = rbase + r;
        if (ebase + row < nEdges) {
          int dn = sDst[wid][row];
          atomicAdd(&aggr[(long long)dn * 64 + col], acc[r]);
        }
      }
    }
  }
}

// ---------------------------------------------------------------------------
// Kernel 2: per-node update MLP
//   out = ReLU(cat[x, aggr] @ U1 + ub1) @ U2 + ub2
// One wave handles tiles of 16 nodes (grid-stride).
// ---------------------------------------------------------------------------
__global__ __launch_bounds__(BLOCK_THREADS) void node_update_kernel(
    const float* __restrict__ x, const float* __restrict__ aggr,
    const float* __restrict__ U1, const float* __restrict__ ub1,
    const float* __restrict__ U2, const float* __restrict__ ub2,
    float* __restrict__ out, int nNodes) {
  __shared__ __align__(32) _Float16 sU1T[64 * 128];
  __shared__ __align__(32) _Float16 sU2T[64 * 64];
  __shared__ float sB1[64];
  __shared__ float sB2[64];
  __shared__ __align__(32) _Float16 sIn[WAVES_PER_BLOCK][16 * 128];
  __shared__ __align__(32) _Float16 sH[WAVES_PER_BLOCK][16 * 64];

  const int tid = threadIdx.x;
  stage_weightT(sU1T, U1, 128, tid);
  stage_weightT(sU2T, U2, 64, tid);
  if (tid < 64) {
    sB1[tid] = ub1[tid];
    sB2[tid] = ub2[tid];
  }
  __syncthreads();

  const int lane = tid & 31;
  const int wid  = tid >> 5;
  const int nTiles = (nNodes + 15) >> 4;

  for (int tile = blockIdx.x * WAVES_PER_BLOCK + wid; tile < nTiles;
       tile += gridDim.x * WAVES_PER_BLOCK) {
    const int nbase = tile << 4;

    // stage concat[x[node], aggr[node]] (16 x 128) fp16
    {
      int nl   = lane >> 1;
      int part = lane & 1;  // 0 -> x cols 0..63, 1 -> aggr cols 64..127
      int node = nbase + nl;
      if (node >= nNodes) node = nNodes - 1;
      const float* srcRow = part ? (aggr + (long long)node * 64)
                                 : (x + (long long)node * 64);
      const float4* xr = (const float4*)srcRow;
      _Float16* drow = &sIn[wid][nl * 128 + part * 64];
#pragma unroll
      for (int j = 0; j < 8; ++j) {
        *(v8h*)(drow + j * 8) = pack8h(xr[2 * j], xr[2 * j + 1]);
      }
    }
    asm volatile("s_wait_dscnt 0" ::: "memory");

    // layer 1
#pragma unroll
    for (int nb = 0; nb < 4; ++nb) {
      float bias = sB1[nb * 16 + (lane & 15)];
      v8f acc = {bias, bias, bias, bias, bias, bias, bias, bias};
#pragma unroll
      for (int kb = 0; kb < 4; ++kb) {
        v16h a = load_frag_a(&sIn[wid][0], 128, lane, kb);
        v16h b = load_frag_b(sU1T, 128, lane, kb, nb);
        acc = wmma_f16(a, b, acc);
      }
      int col = lane & 15, rbase = (lane >> 4) * 8;
#pragma unroll
      for (int r = 0; r < 8; ++r) {
        float v = acc[r] > 0.0f ? acc[r] : 0.0f;
        sH[wid][(rbase + r) * 64 + nb * 16 + col] = (_Float16)v;
      }
    }
    asm volatile("s_wait_dscnt 0" ::: "memory");

    // layer 2 -> direct store
#pragma unroll
    for (int nb = 0; nb < 4; ++nb) {
      float bias = sB2[nb * 16 + (lane & 15)];
      v8f acc = {bias, bias, bias, bias, bias, bias, bias, bias};
#pragma unroll
      for (int kb = 0; kb < 2; ++kb) {
        v16h a = load_frag_a(&sH[wid][0], 64, lane, kb);
        v16h b = load_frag_b(sU2T, 64, lane, kb, nb);
        acc = wmma_f16(a, b, acc);
      }
      int col = nb * 16 + (lane & 15), rbase = (lane >> 4) * 8;
#pragma unroll
      for (int r = 0; r < 8; ++r) {
        int node = nbase + rbase + r;
        if (node < nNodes) out[(long long)node * 64 + col] = acc[r];
      }
    }
  }
}

// ---------------------------------------------------------------------------
// Host-side launcher
// ---------------------------------------------------------------------------
extern "C" void kernel_launch(void* const* d_in, const int* in_sizes, int n_in,
                              void* d_out, int out_size, void* d_ws,
                              size_t ws_size, hipStream_t stream) {
  const float* x   = (const float*)d_in[0];
  const int*   ei  = (const int*)d_in[1];
  const float* W1  = (const float*)d_in[2];
  const float* b1  = (const float*)d_in[3];
  const float* W2  = (const float*)d_in[4];
  const float* b2  = (const float*)d_in[5];
  const float* U1  = (const float*)d_in[6];
  const float* ub1 = (const float*)d_in[7];
  const float* U2  = (const float*)d_in[8];
  const float* ub2 = (const float*)d_in[9];
  float* out  = (float*)d_out;
  float* aggr = (float*)d_ws;  // [nNodes x 64] f32 scratch

  const int d      = 64;
  const int nNodes = in_sizes[0] / d;  // 50000
  const int nEdges = in_sizes[1] / 2;  // 800000

  // 0) zero aggregation buffer (harness poisons workspace)
  {
    long long n = (long long)nNodes * d;
    int blocks = (int)((n + 255) / 256);
    zero_f32_kernel<<<blocks, 256, 0, stream>>>(aggr, n);
  }
  // 1) edge messages + scatter-add (~4 tiles per wave to amortize weight LDS)
  {
    int nTiles = (nEdges + 15) / 16;
    int blocks = (nTiles + 4 * WAVES_PER_BLOCK - 1) / (4 * WAVES_PER_BLOCK);
    edge_msg_kernel<<<blocks, BLOCK_THREADS, 0, stream>>>(x, ei, W1, b1, W2, b2,
                                                          aggr, nEdges);
  }
  // 2) node update
  {
    int nTiles = (nNodes + 15) / 16;
    int blocks = (nTiles + WAVES_PER_BLOCK - 1) / WAVES_PER_BLOCK;
    node_update_kernel<<<blocks, BLOCK_THREADS, 0, stream>>>(
        x, aggr, U1, ub1, U2, ub2, out, nNodes);
  }
}
